// MyLSTM_23656679866589
// MI455X (gfx1250) — matrix operand order repaired and checked by reading, not compiled, not run on hardware
//
#include <hip/hip_runtime.h>

typedef __attribute__((ext_vector_type(16))) _Float16 v16h;
typedef __attribute__((ext_vector_type(8)))  _Float16 v8h;
typedef __attribute__((ext_vector_type(8)))  float    v8f;
typedef __attribute__((ext_vector_type(4)))  float    v4f;

#define B_SZ   1024
#define T_SZ   2048
#define D_IN   64
#define EMB    24
#define UNITS  128
#define GIN    152     // 24 + 128
#define KPAD   160     // GIN padded to multiple of 32
#define NGATE  512     // 4 gates * 128 units

// workspace layout (bytes)
#define OFF_WTG 0                       // transposed gate weights: [512][160] f16 = 160 KB
#define OFF_WTE (NGATE * KPAD * 2)      // transposed emb weights:  [32][64]  f16 = 4 KB
#define OFF_E   (256 * 1024)            // e buffer, time-major: [T][B][24] f16 = 96 MB

// fast activations: hardware TRANS ops instead of IEEE-divide expansions
__device__ __forceinline__ float fast_tanh(float x) {
#if __has_builtin(__builtin_amdgcn_tanhf)
    return __builtin_amdgcn_tanhf(x);          // v_tanh_f32 (gfx1250 TRANS)
#else
    float e = __expf(2.0f * x);
    return 1.0f - 2.0f * __builtin_amdgcn_rcpf(e + 1.0f);
#endif
}
__device__ __forceinline__ float fast_sigmoid(float x) {
    return __builtin_amdgcn_rcpf(1.0f + __expf(-x));   // v_exp_f32 + v_rcp_f32
}

// ---------------------------------------------------------------------------
// Kernel 1: pack/convert weights.
//   wtg[c][r] = f16( W_{gate}[r][c%128] ), gate = c/128, rows >= 152 zero-pad.
//   wte[c][r] = f16( W_emb[r][c] ), cols >= 24 zero-pad.
// ---------------------------------------------------------------------------
__global__ void prep_weights(const float* __restrict__ W_emb,
                             const float* __restrict__ Wf, const float* __restrict__ Wi,
                             const float* __restrict__ Wg, const float* __restrict__ Wo,
                             _Float16* __restrict__ wtg, _Float16* __restrict__ wte) {
    int idx = blockIdx.x * blockDim.x + threadIdx.x;
    const int NG = NGATE * KPAD;  // 81920
    if (idx < NG) {
        int c = idx / KPAD, r = idx % KPAD;
        const float* Ws = (c < 128) ? Wf : (c < 256) ? Wi : (c < 384) ? Wg : Wo;
        float v = (r < GIN) ? Ws[r * UNITS + (c & 127)] : 0.0f;
        wtg[idx] = (_Float16)v;
    } else if (idx < NG + 32 * D_IN) {
        int j = idx - NG;
        int c = j / D_IN, r = j % D_IN;
        float v = (c < EMB) ? W_emb[r * EMB + c] : 0.0f;
        wte[j] = (_Float16)v;
    }
}

// ---------------------------------------------------------------------------
// Kernel 2: embedding  e[t][b][0..23] = sigmoid(x[b][t][:] @ W_emb + b_emb)
// WMMA tiles: M=16 rows of the flattened [B*T, 64] matrix, K=64 (2 chunks),
// N=24 padded to 32 (2 N-tiles). Output written time-major for the scan.
// ---------------------------------------------------------------------------
__global__ void embed_kernel(const float* __restrict__ x,
                             const float* __restrict__ b_emb,
                             const _Float16* __restrict__ wte,
                             _Float16* __restrict__ e_ws) {
    const int lane = threadIdx.x & 31;
    const int hi   = lane >> 4;
    const int li   = lane & 15;
    const int waveId = (blockIdx.x * blockDim.x + threadIdx.x) >> 5;
    const int nWaves = (gridDim.x * blockDim.x) >> 5;

    // B fragments (kept in VGPRs across all tiles): bfrag[kchunk][ntile]
    v16h bfrag[2][2];
#pragma unroll
    for (int k = 0; k < 2; ++k)
#pragma unroll
        for (int nt = 0; nt < 2; ++nt) {
            const _Float16* p = wte + (nt * 16 + li) * D_IN + k * 32 + hi * 16;
            v8h lo = *(const v8h*)p;
            v8h hh = *(const v8h*)(p + 8);
            bfrag[k][nt] = __builtin_shufflevector(lo, hh, 0,1,2,3,4,5,6,7,8,9,10,11,12,13,14,15);
        }

    // loop-invariant bias values for this lane's output columns
    const float bias0 = b_emb[li];
    const float bias1 = (li < 8) ? b_emb[16 + li] : 0.0f;

    const int nTiles = (B_SZ * T_SZ) / 16;  // 131072
    for (int tile = waveId; tile < nTiles; tile += nWaves) {
        const int rowBase = tile * 16;
        const float* xr = x + (size_t)(rowBase + li) * D_IN;
        v8f acc[2] = {{}, {}};
#pragma unroll
        for (int k = 0; k < 2; ++k) {
            // A fragment (16x32 f16): lane<16 -> K{0-15}, lane>=16 -> K{16-31}
            const int base = k * 32 + hi * 16;
            v4f f0 = *(const v4f*)(xr + base);
            v4f f1 = *(const v4f*)(xr + base + 4);
            v4f f2 = *(const v4f*)(xr + base + 8);
            v4f f3 = *(const v4f*)(xr + base + 12);
            v16h a;
#pragma unroll
            for (int j = 0; j < 4; ++j) {
                a[j]      = (_Float16)f0[j];
                a[4 + j]  = (_Float16)f1[j];
                a[8 + j]  = (_Float16)f2[j];
                a[12 + j] = (_Float16)f3[j];
            }
            acc[0] = __builtin_amdgcn_wmma_f32_16x16x32_f16(false, a, false, bfrag[k][0],
                                                            (short)0, acc[0], false, false);
            acc[1] = __builtin_amdgcn_wmma_f32_16x16x32_f16(false, a, false, bfrag[k][1],
                                                            (short)0, acc[1], false, false);
        }
        // epilogue: bias + sigmoid, store f16 time-major
#pragma unroll
        for (int i = 0; i < 8; ++i) {
            int row = i + 8 * hi;
            size_t r = (size_t)rowBase + row;
            int t = (int)(r & (T_SZ - 1));
            int b = (int)(r >> 11);
            _Float16* dst = e_ws + ((size_t)t * B_SZ + b) * EMB;
            dst[li] = (_Float16)fast_sigmoid(acc[0][i] + bias0);
            if (li < 8) dst[16 + li] = (_Float16)fast_sigmoid(acc[1][i] + bias1);
        }
    }
}

// ---------------------------------------------------------------------------
// Kernel 3: persistent recurrent scan. One block = 16 batch rows, 16 waves.
// Each wave owns 32 gate columns (2 N-tiles); weight B-fragments live in
// VGPRs for all 2048 steps. z = [e_t | h] staged in LDS as f16, gate
// activations staged in LDS as f32, c-state in registers. e is register
// double-buffered: the load for step t+1 is issued during step t compute.
// ---------------------------------------------------------------------------
__global__ void __launch_bounds__(512) lstm_kernel(
    const _Float16* __restrict__ e_ws, const _Float16* __restrict__ wtg,
    const float* __restrict__ bf, const float* __restrict__ bi,
    const float* __restrict__ bg, const float* __restrict__ bo,
    const float* __restrict__ W_out, const float* __restrict__ b_out,
    float* __restrict__ out) {
    __shared__ _Float16 zsh[16 * KPAD];     // [row][0..23]=e_t, [24..151]=h, rest pad
    __shared__ float    gatesh[16 * NGATE]; // activated gate values, f i g o blocks

    const int tid  = threadIdx.x;
    const int lane = tid & 31;
    const int w    = tid >> 5;   // wave id 0..15
    const int hi   = lane >> 4;
    const int li   = lane & 15;
    const int b0   = blockIdx.x * 16;

    // zero z (h part and pads)
    for (int i = tid; i < 16 * KPAD; i += blockDim.x) zsh[i] = (_Float16)0.0f;

    // preload this wave's weight B-fragments: bw[ntile][kchunk] (10 x 8 VGPRs)
    v16h bw[2][5];
#pragma unroll
    for (int j = 0; j < 2; ++j) {
        const int ntg = 2 * w + j;
#pragma unroll
        for (int k = 0; k < 5; ++k) {
            const _Float16* p = wtg + (ntg * 16 + li) * KPAD + k * 32 + hi * 16;
            v8h lo = *(const v8h*)p;
            v8h hh = *(const v8h*)(p + 8);
            bw[j][k] = __builtin_shufflevector(lo, hh, 0,1,2,3,4,5,6,7,8,9,10,11,12,13,14,15);
        }
    }

    // loop-invariant epilogue constants (gate id, bias, LDS column) per N-tile
    const int   ntg0 = 2 * w, ntg1 = 2 * w + 1;
    const int   gate0 = ntg0 >> 3, gate1 = ntg1 >> 3;     // 0=f 1=i 2=g 3=o
    const float* bp0 = (gate0 == 0) ? bf : (gate0 == 1) ? bi : (gate0 == 2) ? bg : bo;
    const float* bp1 = (gate1 == 0) ? bf : (gate1 == 1) ? bi : (gate1 == 2) ? bg : bo;
    const float bias0 = bp0[(ntg0 & 7) * 16 + li];
    const float bias1 = bp1[(ntg1 & 7) * 16 + li];
    const bool  isg0 = (gate0 == 2), isg1 = (gate1 == 2);
    const int   col0 = ntg0 * 16 + li, col1 = ntg1 * 16 + li;

    // c-state: thread tid owns 4 consecutive (row,unit) pairs
    const int p0    = tid * 4;
    const int crow  = p0 >> 7;
    const int cunit = p0 & 127;
    float c0 = 0.0f, c1 = 0.0f, c2 = 0.0f, c3 = 0.0f;

    // e_t register double-buffer: 96 threads x 8 B = 16 rows x 24 f16
    const int erow = (tid * 4) / 24, ecol = (tid * 4) % 24;
    unsigned long long ereg = 0ULL;
    if (tid < 96)
        ereg = ((const unsigned long long*)(e_ws + (size_t)b0 * EMB))[tid];

    __syncthreads();

    for (int t = 0; t < T_SZ; ++t) {
        // commit pre-loaded e_t into LDS (cheap: one ds_store_b64 per thread)
        if (tid < 96)
            *(unsigned long long*)(&zsh[erow * KPAD + ecol]) = ereg;
        if (lane == 0 && t + 2 < T_SZ)
            __builtin_prefetch(e_ws + ((size_t)(t + 2) * B_SZ + b0) * EMB, 0, 0);
        __syncthreads();  // e_t staged; h from previous combine visible

        // issue the global load for e_{t+1} now; latency overlaps the GEMM
        if (tid < 96 && t + 1 < T_SZ)
            ereg = ((const unsigned long long*)(e_ws + ((size_t)(t + 1) * B_SZ + b0) * EMB))[tid];

        // 16x512 gate GEMM over K=160: 2 tiles x 5 chunks per wave
        v8f acc[2] = {{}, {}};
#pragma unroll
        for (int k = 0; k < 5; ++k) {
            const _Float16* p = &zsh[li * KPAD + k * 32 + hi * 16];
            v8h lo = *(const v8h*)p;
            v8h hh = *(const v8h*)(p + 8);
            v16h a = __builtin_shufflevector(lo, hh, 0,1,2,3,4,5,6,7,8,9,10,11,12,13,14,15);
            acc[0] = __builtin_amdgcn_wmma_f32_16x16x32_f16(false, a, false, bw[0][k],
                                                            (short)0, acc[0], false, false);
            acc[1] = __builtin_amdgcn_wmma_f32_16x16x32_f16(false, a, false, bw[1][k],
                                                            (short)0, acc[1], false, false);
        }

        // bias + activation into LDS gate buffer (v_tanh_f32 / v_exp+v_rcp)
#pragma unroll
        for (int i = 0; i < 8; ++i) {
            const int row = i + 8 * hi;
            float v0 = acc[0][i] + bias0;
            float v1 = acc[1][i] + bias1;
            v0 = isg0 ? fast_tanh(v0) : fast_sigmoid(v0);
            v1 = isg1 ? fast_tanh(v1) : fast_sigmoid(v1);
            gatesh[row * NGATE + col0] = v0;
            gatesh[row * NGATE + col1] = v1;
        }
        __syncthreads();  // gate buffer complete

        // combine: c = c*f + i*g ; h = tanh(c)*o ; write h back as f16
        {
            const float* gp = &gatesh[crow * NGATE + cunit];
            v4f fv = *(const v4f*)(gp);
            v4f iv = *(const v4f*)(gp + 128);
            v4f gv = *(const v4f*)(gp + 256);
            v4f ov = *(const v4f*)(gp + 384);
            c0 = c0 * fv[0] + iv[0] * gv[0];
            c1 = c1 * fv[1] + iv[1] * gv[1];
            c2 = c2 * fv[2] + iv[2] * gv[2];
            c3 = c3 * fv[3] + iv[3] * gv[3];
            _Float16* hz = &zsh[crow * KPAD + 24 + cunit];
            hz[0] = (_Float16)(fast_tanh(c0) * ov[0]);
            hz[1] = (_Float16)(fast_tanh(c1) * ov[1]);
            hz[2] = (_Float16)(fast_tanh(c2) * ov[2]);
            hz[3] = (_Float16)(fast_tanh(c3) * ov[3]);
        }
        // no barrier needed here: next iter's e-writes touch disjoint LDS,
        // and the next __syncthreads() orders h-writes vs. A-fragment reads.
    }

    __syncthreads();
    // output head: out[b] = sigmoid(h_last . W_out + b_out)
    if (tid < 16) {
        const _Float16* hrow = &zsh[tid * KPAD + 24];
        float s = 0.0f;
#pragma unroll 8
        for (int u = 0; u < UNITS; ++u) s += (float)hrow[u] * W_out[u];
        out[b0 + tid] = fast_sigmoid(s + b_out[0]);
    }
}

// ---------------------------------------------------------------------------
extern "C" void kernel_launch(void* const* d_in, const int* in_sizes, int n_in,
                              void* d_out, int out_size, void* d_ws, size_t ws_size,
                              hipStream_t stream) {
    const float* x     = (const float*)d_in[0];
    const float* W_emb = (const float*)d_in[1];
    const float* b_emb = (const float*)d_in[2];
    const float* Wf    = (const float*)d_in[3];
    const float* bf    = (const float*)d_in[4];
    const float* Wi    = (const float*)d_in[5];
    const float* bi    = (const float*)d_in[6];
    const float* Wg    = (const float*)d_in[7];
    const float* bg    = (const float*)d_in[8];
    const float* Wo    = (const float*)d_in[9];
    const float* bo    = (const float*)d_in[10];
    const float* W_out = (const float*)d_in[11];
    const float* b_out = (const float*)d_in[12];
    float* out = (float*)d_out;

    char* ws = (char*)d_ws;
    _Float16* wtg  = (_Float16*)(ws + OFF_WTG);
    _Float16* wte  = (_Float16*)(ws + OFF_WTE);
    _Float16* e_ws = (_Float16*)(ws + OFF_E);

    const int prepN = NGATE * KPAD + 32 * D_IN;
    prep_weights<<<(prepN + 255) / 256, 256, 0, stream>>>(W_emb, Wf, Wi, Wg, Wo, wtg, wte);
    embed_kernel<<<512, 256, 0, stream>>>(x, b_emb, wte, e_ws);
    lstm_kernel<<<B_SZ / 16, 512, 0, stream>>>(e_ws, wtg, bf, bi, bg, bo,
                                               W_out, b_out, out);
}
